// GCNEncoderWithFeatures_3874060501838
// MI455X (gfx1250) — matrix operand order repaired and checked by reading, not compiled, not run on hardware
//
#include <hip/hip_runtime.h>

// ---------------------------------------------------------------------------
// GCN encoder for MI455X (gfx1250, wave32, WMMA + Tensor Data Mover).
// GEMMs: TDM stages the shared 16xK A-panel into LDS (padded against bank
// conflicts), 13 waves/block each compute one 16x16 tile with
// v_wmma_f32_16x16x32_f16 (f16 in, f32 accumulate). K/S are compile-time so
// all LDS/global fragment addresses fold to immediate offsets.
// Aggregation: deterministic CSR segment-sum (dst sorted) -> no atomics.
// ---------------------------------------------------------------------------

typedef __attribute__((ext_vector_type(16))) _Float16     v16h;
typedef __attribute__((ext_vector_type(8)))  _Float16     v8h;
typedef __attribute__((ext_vector_type(8)))  float        v8f;
typedef __attribute__((ext_vector_type(4)))  float        v4f;
typedef __attribute__((ext_vector_type(4)))  unsigned int v4u;
typedef __attribute__((ext_vector_type(8)))  int          v8i;
typedef __attribute__((ext_vector_type(4)))  int          v4i;

#define N_NODES  100000
#define N_FEAT   512
#define HIDDEN   200
#define HPAD16   208             // hidden padded to multiple of 16 (N-tile dim)
#define HPAD32   224             // hidden padded to multiple of 32 (K dim for WMMA)
#define COL_TILES (HPAD16 / 16)  // 13
#define ROW_TILES (N_NODES / 16) // 6250
#define GEMM_THREADS (COL_TILES * 32) // 416: one wave per column tile
// LDS A-panel: 16 rows, K + pad halfs per row; max K=512 -> stride 520 halfs
#define SA_HALFS (16 * 520)

// ---------------------------------------------------------------------------
// helpers
// ---------------------------------------------------------------------------
__device__ __forceinline__ v16h cat16(v8h lo, v8h hi) {
  v16h r;
#pragma unroll
  for (int i = 0; i < 8; ++i) { r[i] = lo[i]; r[i + 8] = hi[i]; }
  return r;
}

// TDM: DMA a 16-row x K-halfs tile (row stride As halfs) from global into LDS.
// D# per CDNA5 ISA ch.8: group0 = {flags, lds_addr, gaddr_lo, gaddr_hi|type},
// group1 = dims/strides. Padding: every 2^(S-1) DWORDs stored, insert 4 DWORDs
// (16B) -> LDS row stride = K + ((K>>S)<<3) halfs, conflict-free ds_load_b128.
__device__ __forceinline__ void tdm_load_A(const _Float16* gsrc, unsigned ldsByteOff,
                                           int K, int As, int S) {
  unsigned long long ga = (unsigned long long)(size_t)gsrc;
  v4u g0;
  g0[0] = 1u;                                        // count=1, user mode
  g0[1] = ldsByteOff;                                // lds_addr (bytes)
  g0[2] = (unsigned)(ga & 0xffffffffu);              // global_addr[31:0]
  g0[3] = (unsigned)((ga >> 32) & 0x01ffffffu)       // global_addr[56:32]
        | (2u << 30);                                // type = 2 ("image")
  v8i g1;
  unsigned padIntervalCode = (unsigned)(S - 2);      // 2^(c+1) DW = 2^(S-1) DW
  g1[0] = (int)((1u << 16)                           // data_size = 1 -> 2 bytes
              | (1u << 20)                           // pad_enable
              | (padIntervalCode << 22)              // pad_interval
              | (3u << 25));                         // pad_amount = 4 DW (16B)
  g1[1] = (int)(((unsigned)K & 0xffffu) << 16);      // tensor_dim0[15:0]
  g1[2] = (int)((((unsigned)K >> 16) & 0xffffu)      // tensor_dim0[31:16]
              | (16u << 16));                        // tensor_dim1 = 16 rows
  g1[3] = (int)(((unsigned)K & 0xffffu) << 16);      // dim1 hi=0, tile_dim0=K
  g1[4] = (int)16u;                                  // tile_dim1=16, tile_dim2=0
  g1[5] = (int)As;                                   // tensor_dim0_stride[31:0]
  g1[6] = 0;                                         // stride0 hi, stride1 lo
  g1[7] = 0;
  v4i z4 = {};
#if __clang_major__ >= 23
  v8i z8 = {};
  __builtin_amdgcn_tensor_load_to_lds(g0, g1, z4, z4, z8, 0);
#else
  __builtin_amdgcn_tensor_load_to_lds(g0, g1, z4, z4, 0);
#endif
}

// One K-panel phase: wave 0 DMAs A panel to LDS, then each wave accumulates
// its 16x16 tile. S: log2 of pad interval in halfs (9 for K=512, 5 for K=224).
// Note: for kc a multiple of 32, (g*8+kc)>>S and (g*8+kc+16)>>S both equal
// kc>>S (since g*8+16+7 < 32 <= 2^S), so pad-adjusted offsets are lane-
// independent compile-time immediates off a per-lane base.
template <int K, int S>
__device__ __forceinline__ v8f gemm_phase(const _Float16* __restrict__ A, int As,
                                          const _Float16* __restrict__ Bt,
                                          _Float16* sA,
                                          int tm, int tn, int g, int l, int wave,
                                          v8f acc) {
  __syncthreads();                       // previous users of sA are done
  if (wave == 0) {
    tdm_load_A(A + (size_t)tm * 16 * As, (unsigned)(size_t)sA, K, As, S);
    __builtin_amdgcn_s_wait_tensorcnt((short)0);
  }
  __syncthreads();                       // LDS panel visible to all waves
  constexpr int rowStride = K + ((K >> S) << 3);
  const _Float16* sAp  = sA + l * rowStride + g * 8;           // per-lane base
  const _Float16* Bcol = Bt + (size_t)(tn * 16 + l) * K + g * 8;
#pragma unroll
  for (int kc = 0; kc < K; kc += 32) {
    const int eff = kc + ((kc >> S) << 3);   // compile-time after unroll
    v8h a0 = *(const v8h*)(sAp + eff);       // K = kc + g*8 + 0..7
    v8h a1 = *(const v8h*)(sAp + eff + 16);  // K = kc + g*8 + 16..23
    v8h b0 = *(const v8h*)(Bcol + kc);       // K = kc + g*8 + 0..7
    v8h b1 = *(const v8h*)(Bcol + kc + 16);  // K = kc + g*8 + 16..23
    acc = __builtin_amdgcn_wmma_f32_16x16x32_f16(
        /*neg_a=*/false, cat16(a0, a1),
        /*neg_b=*/false, cat16(b0, b1),
        /*c_mod=*/(short)0, acc, /*reuse_a=*/false, /*reuse_b=*/false);
  }
  return acc;
}

// ---------------------------------------------------------------------------
// conversion kernels
// ---------------------------------------------------------------------------
__global__ void __launch_bounds__(256)
f32_to_f16_rows(const float* __restrict__ src, int srcStride, int usedCols,
                _Float16* __restrict__ dst, int dstStride, long total) {
  long idx = (long)blockIdx.x * 256 + threadIdx.x;
  if (idx >= total) return;
  long r = idx / dstStride;
  int  c = (int)(idx % dstStride);
  float v = (c < usedCols) ? src[r * (size_t)srcStride + c] : 0.f;
  dst[idx] = (_Float16)v;
}

__global__ void __launch_bounds__(256)
weight_to_f16_t(const float* __restrict__ src, int srcCols, int rowOff,
                int Kused, int Kpad, _Float16* __restrict__ dst, int total) {
  int idx = blockIdx.x * 256 + threadIdx.x;
  if (idx >= total) return;
  int c = idx / Kpad;
  int k = idx % Kpad;
  float v = (k < Kused && c < srcCols) ? src[(size_t)(rowOff + k) * srcCols + c] : 0.f;
  dst[idx] = (_Float16)v;
}

// ---------------------------------------------------------------------------
// GEMM kernels: grid = ROW_TILES blocks, 13 waves = 13 column tiles each
// ---------------------------------------------------------------------------
template <int K, int S>
__global__ void __launch_bounds__(GEMM_THREADS)
gemm_xw_kernel(const _Float16* __restrict__ A, int As,
               const _Float16* __restrict__ Bt,
               float* __restrict__ C, int Cs) {
  __shared__ __align__(16) _Float16 sA[SA_HALFS];
  int wave = threadIdx.x >> 5, lane = threadIdx.x & 31;
  int tm = blockIdx.x, tn = wave;
  int g = lane >> 4, l = lane & 15;
  v8f acc = {};
  acc = gemm_phase<K, S>(A, As, Bt, sA, tm, tn, g, l, wave, acc);
  // C/D layout: VGPR r -> row tm*16 + r + g*8, col tn*16 + l
  float* Cp = C + (size_t)(tm * 16 + g * 8) * Cs + tn * 16 + l;
#pragma unroll
  for (int r = 0; r < 8; ++r) Cp[(size_t)r * Cs] = acc[r];
}

// out = relu(A1 @ B1t + A2 @ B2t)  (fused "concat" GEMM)
template <int K1, int S1, int K2, int S2>
__global__ void __launch_bounds__(GEMM_THREADS)
gemm_th_kernel(const _Float16* __restrict__ A1, int As1,
               const _Float16* __restrict__ B1t,
               const _Float16* __restrict__ A2, int As2,
               const _Float16* __restrict__ B2t,
               float* __restrict__ out, _Float16* __restrict__ hh) {
  __shared__ __align__(16) _Float16 sA[SA_HALFS];
  int wave = threadIdx.x >> 5, lane = threadIdx.x & 31;
  int tm = blockIdx.x, tn = wave;
  int g = lane >> 4, l = lane & 15;
  v8f acc = {};
  acc = gemm_phase<K1, S1>(A1, As1, B1t, sA, tm, tn, g, l, wave, acc); // agg part
  acc = gemm_phase<K2, S2>(A2, As2, B2t, sA, tm, tn, g, l, wave, acc); // feature part
  int col  = tn * 16 + l;
  int row0 = tm * 16 + g * 8;
#pragma unroll
  for (int r = 0; r < 8; ++r) {
    float v = acc[r] > 0.f ? acc[r] : 0.f;  // relu
    if (col < HIDDEN) out[(size_t)(row0 + r) * HIDDEN + col] = v;
  }
  if (hh != nullptr) {
#pragma unroll
    for (int r = 0; r < 8; ++r) {
      float v = acc[r] > 0.f ? acc[r] : 0.f;
      hh[(size_t)(row0 + r) * HPAD32 + col] = (col < HIDDEN) ? (_Float16)v : (_Float16)0.f;
    }
  }
}

// ---------------------------------------------------------------------------
// aggregation: rowptr from sorted dst, then per-node segment sum (deterministic)
// ---------------------------------------------------------------------------
__global__ void __launch_bounds__(256)
build_rowptr(const int* __restrict__ dst, int nEdges, int* __restrict__ rowptr, int nNodes) {
  long e = (long)blockIdx.x * 256 + threadIdx.x;
  if (e >= nEdges) return;
  int d1 = dst[e];
  int d0 = (e == 0) ? -1 : dst[e - 1];
  for (int n = d0 + 1; n <= d1; ++n) rowptr[n] = (int)e;
  if (e == nEdges - 1)
    for (int n = d1 + 1; n <= nNodes; ++n) rowptr[n] = nEdges;
}

// agg[n, 0:200] = sum_{e in [rowptr[n],rowptr[n+1])} ew[e] * xw[src[e], 0:200]
__global__ void __launch_bounds__(256)
spmm_csr_kernel(const float* __restrict__ xw,
                const int* __restrict__ rowptr,
                const int* __restrict__ src, const float* __restrict__ ew,
                float* __restrict__ agg, int nNodes) {
  int node = blockIdx.x * 4 + threadIdx.y;
  if (node >= nNodes) return;
  int cg = threadIdx.x;
  if (cg >= HIDDEN / 4) return;               // 50 column groups of float4
  int e0 = rowptr[node], e1 = rowptr[node + 1];
  v4f acc = {};
  for (int e = e0; e < e1; ++e) {
    int   s  = src[e];                        // wave-uniform address
    float wv = ew[e];
    acc += wv * *(const v4f*)(xw + (size_t)s * HPAD16 + cg * 4);
  }
  *(v4f*)(agg + (size_t)node * HPAD16 + cg * 4) = acc;
}

// ---------------------------------------------------------------------------
// launch
// ---------------------------------------------------------------------------
extern "C" void kernel_launch(void* const* d_in, const int* in_sizes, int n_in,
                              void* d_out, int out_size, void* d_ws, size_t ws_size,
                              hipStream_t stream) {
  const float* features = (const float*)d_in[0];
  const int*   src      = (const int*)d_in[1];
  const int*   dst      = (const int*)d_in[2];
  const float* ew       = (const float*)d_in[3];
  const float* W_gc1    = (const float*)d_in[4];
  const float* W_gc2    = (const float*)d_in[5];
  const float* W_th     = (const float*)d_in[6];
  const float* W_th1    = (const float*)d_in[7];
  const int nEdges = in_sizes[1];

  float* out1 = (float*)d_out;
  float* out2 = out1 + (size_t)N_NODES * HIDDEN;

  // workspace carve-up (all offsets 256B-aligned by construction)
  char* w = (char*)d_ws;
  _Float16* Xh    = (_Float16*)w; w += (size_t)N_NODES * N_FEAT * 2;   // 102.4 MB
  _Float16* W1t   = (_Float16*)w; w += (size_t)HPAD16 * N_FEAT * 2;
  _Float16* W2t   = (_Float16*)w; w += (size_t)HPAD16 * HPAD32 * 2;
  _Float16* WthA  = (_Float16*)w; w += (size_t)HPAD16 * HPAD32 * 2;
  _Float16* WthB  = (_Float16*)w; w += (size_t)HPAD16 * N_FEAT * 2;
  _Float16* Wth1A = (_Float16*)w; w += (size_t)HPAD16 * HPAD32 * 2;
  _Float16* Wth1B = (_Float16*)w; w += (size_t)HPAD16 * N_FEAT * 2;
  float*    xw    = (float*)w;    w += (size_t)N_NODES * HPAD16 * 4;   // 83.2 MB
  float*    agg   = (float*)w;    w += (size_t)N_NODES * HPAD16 * 4;   // 83.2 MB
  _Float16* aggh  = (_Float16*)w; w += (size_t)N_NODES * HPAD32 * 2;   // 44.8 MB
  _Float16* h1h   = (_Float16*)w; w += (size_t)N_NODES * HPAD32 * 2;   // 44.8 MB
  int*      rowptr= (int*)w;      w += (((size_t)(N_NODES + 1) * 4 + 255) & ~(size_t)255);

  // --- per-launch conversions ---
  { long total = (long)N_NODES * N_FEAT;
    f32_to_f16_rows<<<(int)((total + 255) / 256), 256, 0, stream>>>(
        features, N_FEAT, N_FEAT, Xh, N_FEAT, total); }
  { int total = HPAD16 * N_FEAT;
    weight_to_f16_t<<<(total + 255) / 256, 256, 0, stream>>>(W_gc1, HIDDEN, 0,      N_FEAT, N_FEAT, W1t,   total);
    weight_to_f16_t<<<(total + 255) / 256, 256, 0, stream>>>(W_th,  HIDDEN, HIDDEN, N_FEAT, N_FEAT, WthB,  total);
    weight_to_f16_t<<<(total + 255) / 256, 256, 0, stream>>>(W_th1, HIDDEN, HIDDEN, N_FEAT, N_FEAT, Wth1B, total); }
  { int total = HPAD16 * HPAD32;
    weight_to_f16_t<<<(total + 255) / 256, 256, 0, stream>>>(W_gc2, HIDDEN, 0, HIDDEN, HPAD32, W2t,   total);
    weight_to_f16_t<<<(total + 255) / 256, 256, 0, stream>>>(W_th,  HIDDEN, 0, HIDDEN, HPAD32, WthA,  total);
    weight_to_f16_t<<<(total + 255) / 256, 256, 0, stream>>>(W_th1, HIDDEN, 0, HIDDEN, HPAD32, Wth1A, total); }

  build_rowptr<<<(nEdges + 255) / 256, 256, 0, stream>>>(dst, nEdges, rowptr, N_NODES);

  const dim3 aggBlk(64, 4);
  const int  aggBlocks = (N_NODES + 3) / 4;

  // --- layer 1 ---
  gemm_xw_kernel<N_FEAT, 9><<<ROW_TILES, GEMM_THREADS, 0, stream>>>(Xh, N_FEAT, W1t, xw, HPAD16);
  spmm_csr_kernel<<<aggBlocks, aggBlk, 0, stream>>>(xw, rowptr, src, ew, agg, N_NODES);
  { long total = (long)N_NODES * HPAD32;
    f32_to_f16_rows<<<(int)((total + 255) / 256), 256, 0, stream>>>(
        agg, HPAD16, HIDDEN, aggh, HPAD32, total); }
  hipMemsetAsync(h1h, 0, (size_t)N_NODES * HPAD32 * 2, stream);  // zero pad cols 208..223
  gemm_th_kernel<HPAD32, 5, N_FEAT, 9><<<ROW_TILES, GEMM_THREADS, 0, stream>>>(
      aggh, HPAD32, WthA, Xh, N_FEAT, WthB, out1, h1h);

  // --- layer 2 ---
  gemm_xw_kernel<HPAD32, 5><<<ROW_TILES, GEMM_THREADS, 0, stream>>>(h1h, HPAD32, W2t, xw, HPAD16);
  spmm_csr_kernel<<<aggBlocks, aggBlk, 0, stream>>>(xw, rowptr, src, ew, agg, N_NODES);
  { long total = (long)N_NODES * HPAD32;
    f32_to_f16_rows<<<(int)((total + 255) / 256), 256, 0, stream>>>(
        agg, HPAD16, HIDDEN, aggh, HPAD32, total); }
  gemm_th_kernel<HPAD32, 5, N_FEAT, 9><<<ROW_TILES, GEMM_THREADS, 0, stream>>>(
      aggh, HPAD32, Wth1A, Xh, N_FEAT, Wth1B, out2, nullptr);
}
